// GAT_37744172597776
// MI455X (gfx1250) — matrix operand order repaired and checked by reading, not compiled, not run on hardware
//
#include <hip/hip_runtime.h>
#include <hip/hip_bf16.h>
#include <math.h>

#define NN      20000          // nodes
#define NE      320000         // edges (without self loops)
#define ETOT    (NE + NN)      // edges + self loops
#define NG      64             // graphs
#define FIN     128
#define HID     32
#define HEADS   8
#define NCLS    40
#define D1      (HEADS * HID)  // 256
#define MAXK    256            // max K of either GEMM (LDS B-tile sizing)

typedef __attribute__((ext_vector_type(16))) _Float16 v16h;
typedef __attribute__((ext_vector_type(8)))  _Float16 h8;
typedef __attribute__((ext_vector_type(8)))  float    v8f;

// ---------------- helpers ----------------
__device__ __forceinline__ unsigned fenc(float f) {           // monotone float->uint
    unsigned u = __float_as_uint(f);
    return (u & 0x80000000u) ? ~u : (u | 0x80000000u);
}
__device__ __forceinline__ float fdec(unsigned e) {
    return (e & 0x80000000u) ? __uint_as_float(e ^ 0x80000000u)
                             : __uint_as_float(~e);
}
__device__ __forceinline__ float lrelu(float v) { return v > 0.f ? v : 0.2f * v; }
__device__ __forceinline__ float eluf (float v) { return v > 0.f ? v : expm1f(v); }

// ---------------- conversion kernels ----------------
__global__ void cvt_f32_f16(const float* __restrict__ in, _Float16* __restrict__ out, int n) {
    int i = blockIdx.x * blockDim.x + threadIdx.x;
    if (i < n) out[i] = (_Float16)in[i];
}
// out[n*K + k] = in[k*N + n]  (transpose + cvt): B stored K-contiguous per column
__global__ void cvt_transpose_f16(const float* __restrict__ in, _Float16* __restrict__ out,
                                  int Ncols, int K) {
    int i = blockIdx.x * blockDim.x + threadIdx.x;
    if (i >= Ncols * K) return;
    int n = i / K, k = i % K;
    out[i] = (_Float16)in[(size_t)k * Ncols + n];
}
__global__ void init_u32(unsigned* __restrict__ p, unsigned v, int n) {
    int i = blockIdx.x * blockDim.x + threadIdx.x;
    if (i < n) p[i] = v;
}

// ---------------- WMMA GEMM: C[M x Nc] = A[M x K] * Bt[Nc x K]^T ----------------
// Block = 8 waves = 8 row-tiles sharing ONE 16-column B tile staged in LDS.
// Each wave: one 16x16 output tile, K stepped by 32 via v_wmma_f32_16x16x32_f16.
__global__ void wmma_gemm_f16(const _Float16* __restrict__ A,
                              const _Float16* __restrict__ Bt,
                              float* __restrict__ C,
                              int K, int Nc, int rowTiles, int rbBlocks) {
    __shared__ _Float16 Bs[16 * MAXK];                 // <= 8 KB of the 320 KB WGP LDS
    int tN = blockIdx.x / rbBlocks;
    int rb = blockIdx.x % rbBlocks;

    // cooperative contiguous copy of B column-tile (16 cols x K halfs) into LDS
    {
        const uint4* src = (const uint4*)(Bt + (size_t)tN * 16 * K);
        uint4*       dst = (uint4*)Bs;
        int nvec = (16 * K) >> 3;                      // 8 halfs per uint4
        for (int i = threadIdx.x; i < nvec; i += blockDim.x) dst[i] = src[i];
    }
    __syncthreads();

    int wave = threadIdx.x >> 5;
    int tM   = rb * 8 + wave;
    if (tM >= rowTiles) return;                        // wave-uniform: EXEC stays all-1s
    int lane = threadIdx.x & 31;
    int kaOff = (lane < 16) ? 0 : 8;                   // ISA 16-bit A-fragment layout
    int kbOff = (lane < 16) ? 0 : 8;                   // ISA 16-bit B-fragment layout
    const _Float16* ap = A  + (size_t)(tM * 16 + (lane & 15)) * K;
    const _Float16* bp = Bs + (lane & 15) * K;         // LDS: ds_load_b128 fragments

    v8f c = {};
    for (int k0 = 0; k0 < K; k0 += 32) {
        h8 a0 = *(const h8*)(ap + k0 + kaOff);
        h8 a1 = *(const h8*)(ap + k0 + 16 + kaOff);
        h8 b0 = *(const h8*)(bp + k0 + kbOff);
        h8 b1 = *(const h8*)(bp + k0 + 16 + kbOff);
        v16h a = __builtin_shufflevector(a0, a1, 0,1,2,3,4,5,6,7,8,9,10,11,12,13,14,15);
        v16h b = __builtin_shufflevector(b0, b1, 0,1,2,3,4,5,6,7,8,9,10,11,12,13,14,15);
        c = __builtin_amdgcn_wmma_f32_16x16x32_f16(false, a, false, b,
                                                   (short)0, c, false, false);
    }
    int col   = tN * 16 + (lane & 15);
    int mBase = tM * 16 + ((lane < 16) ? 0 : 8);       // D layout: VGPR r -> M=r (+8 hi lanes)
    for (int r = 0; r < 8; ++r)
        C[(size_t)(mBase + r) * Nc + col] = c[r];
}

// ---------------- attention scalar dots: a_src/a_dst [N,H] ----------------
__global__ void attn_dot(const float* __restrict__ hfeat,
                         const float* __restrict__ att_s, const float* __restrict__ att_d,
                         float* __restrict__ as, float* __restrict__ ad, int H) {
    int t = blockIdx.x * blockDim.x + threadIdx.x;
    if (t >= NN * H) return;
    int n = t / H, hd = t % H;
    const float* hp = hfeat + (size_t)n * H * HID + hd * HID;
    float s = 0.f, d = 0.f;
    for (int c = 0; c < HID; ++c) {
        float v = hp[c];
        s += v * att_s[hd * HID + c];
        d += v * att_d[hd * HID + c];
    }
    as[t] = s; ad[t] = d;
}

// ---------------- edge passes (self loops appended: ee >= NE -> (i,i)) ----------------
__global__ void edge_max(const int* __restrict__ ei, const float* __restrict__ as,
                         const float* __restrict__ ad, unsigned* __restrict__ emax, int H) {
    int ee = blockIdx.x * blockDim.x + threadIdx.x;
    if (ee >= ETOT) return;
    int s, d;
    if (ee < NE) { s = ei[ee]; d = ei[NE + ee]; } else { s = d = ee - NE; }
    for (int h = 0; h < H; ++h) {
        float e = lrelu(as[s * H + h] + ad[d * H + h]);
        atomicMax(&emax[d * H + h], fenc(e));
    }
}
__global__ void edge_exp(const int* __restrict__ ei, const float* __restrict__ as,
                         const float* __restrict__ ad, const unsigned* __restrict__ emax,
                         float* __restrict__ exbuf, float* __restrict__ denom, int H) {
    int ee = blockIdx.x * blockDim.x + threadIdx.x;
    if (ee >= ETOT) return;
    int s, d;
    if (ee < NE) { s = ei[ee]; d = ei[NE + ee]; } else { s = d = ee - NE; }
    for (int h = 0; h < H; ++h) {
        float e  = lrelu(as[s * H + h] + ad[d * H + h]);
        float ex = __expf(e - fdec(emax[d * H + h]));
        exbuf[(size_t)ee * H + h] = ex;
        atomicAdd(&denom[d * H + h], ex);
    }
}
__global__ void edge_aggr(const int* __restrict__ ei, const float* __restrict__ exbuf,
                          const float* __restrict__ denom, const float* __restrict__ hfeat,
                          float* __restrict__ out, int H) {
    int t = blockIdx.x * blockDim.x + threadIdx.x;
    if (t >= ETOT * H) return;
    int ee = t / H, h = t % H;
    int s, d;
    if (ee < NE) { s = ei[ee]; d = ei[NE + ee]; } else { s = d = ee - NE; }
    const float* hp = hfeat + (size_t)s * H * HID + h * HID;
    float*       op = out   + (size_t)d * H * HID + h * HID;
    __builtin_prefetch(hp, 0, 1);                      // global_prefetch_b8: hide gather latency
    float alpha = exbuf[t] / (denom[d * H + h] + 1e-16f);
    for (int c = 0; c < HID; ++c) atomicAdd(&op[c], alpha * hp[c]);
}

// ---------------- epilogues ----------------
__global__ void epi1(const float* __restrict__ acc, const float* __restrict__ bias,
                     _Float16* __restrict__ h16) {   // bias + ELU -> f16 A-matrix for GEMM2
    int t = blockIdx.x * blockDim.x + threadIdx.x;
    if (t >= NN * D1) return;
    h16[t] = (_Float16)eluf(acc[t] + bias[t % D1]);
}
__global__ void epi2(const float* __restrict__ acc, const float* __restrict__ bias,
                     float* __restrict__ h2) {
    int t = blockIdx.x * blockDim.x + threadIdx.x;
    if (t >= NN * HID) return;
    h2[t] = eluf(acc[t] + bias[t % HID]);
}

// ---------------- global mean pool + FC ----------------
__global__ void pool_sum(const float* __restrict__ h2, const int* __restrict__ batch,
                         float* __restrict__ gsum, float* __restrict__ cnt) {
    int t = blockIdx.x * blockDim.x + threadIdx.x;
    if (t >= NN * HID) return;
    int n = t / HID, c = t % HID;
    int g = batch[n];
    atomicAdd(&gsum[g * HID + c], h2[t]);
    if (c == 0) atomicAdd(&cnt[g], 1.0f);
}
__global__ void fc_out(const float* __restrict__ gsum, const float* __restrict__ cnt,
                       const float* __restrict__ fcw, const float* __restrict__ fcb,
                       float* __restrict__ out) {
    int t = blockIdx.x * blockDim.x + threadIdx.x;
    if (t >= NG * NCLS) return;
    int g = t / NCLS, cl = t % NCLS;
    float inv = 1.0f / fmaxf(cnt[g], 1.0f);
    float acc = fcb[cl];
    for (int c = 0; c < HID; ++c)
        acc += (gsum[g * HID + c] * inv) * fcw[c * NCLS + cl];
    out[t] = acc;
}

// ---------------- launch ----------------
static inline int cdiv(long long a, long long b) { return (int)((a + b - 1) / b); }

extern "C" void kernel_launch(void* const* d_in, const int* in_sizes, int n_in,
                              void* d_out, int out_size, void* d_ws, size_t ws_size,
                              hipStream_t stream) {
    const float* x    = (const float*)d_in[0];
    const int*   ei   = (const int*)  d_in[1];
    const int*   batch= (const int*)  d_in[2];
    const float* W1   = (const float*)d_in[3];
    const float* asr1 = (const float*)d_in[4];
    const float* adr1 = (const float*)d_in[5];
    const float* b1   = (const float*)d_in[6];
    const float* W2   = (const float*)d_in[7];
    const float* asr2 = (const float*)d_in[8];
    const float* adr2 = (const float*)d_in[9];
    const float* b2   = (const float*)d_in[10];
    const float* fcw  = (const float*)d_in[11];
    const float* fcb  = (const float*)d_in[12];
    float* out = (float*)d_out;

    // workspace carve-up (256B aligned slices); total ~80 MB
    char* p = (char*)d_ws;
    auto take = [&](size_t bytes) { char* r = p; p += (bytes + 255) & ~(size_t)255; return r; };
    _Float16* x16    = (_Float16*)take((size_t)NN * FIN * 2);
    _Float16* W1t    = (_Float16*)take((size_t)D1 * FIN * 2);
    _Float16* W2t    = (_Float16*)take((size_t)HID * D1 * 2);
    float*    hfeat1 = (float*)   take((size_t)NN * D1 * 4);
    float*    as1    = (float*)   take((size_t)NN * HEADS * 4);
    float*    ad1    = (float*)   take((size_t)NN * HEADS * 4);
    unsigned* emax1  = (unsigned*)take((size_t)NN * HEADS * 4);
    float*    den1   = (float*)   take((size_t)NN * HEADS * 4);
    float*    ex1    = (float*)   take((size_t)ETOT * HEADS * 4);
    float*    out1   = (float*)   take((size_t)NN * D1 * 4);
    _Float16* h1p16  = (_Float16*)take((size_t)NN * D1 * 2);
    float*    hfeat2 = (float*)   take((size_t)NN * HID * 4);
    float*    as2    = (float*)   take((size_t)NN * 4);
    float*    ad2    = (float*)   take((size_t)NN * 4);
    unsigned* emax2  = (unsigned*)take((size_t)NN * 4);
    float*    den2   = (float*)   take((size_t)NN * 4);
    float*    ex2    = (float*)   take((size_t)ETOT * 4);
    float*    out2   = (float*)   take((size_t)NN * HID * 4);
    float*    h2     = (float*)   take((size_t)NN * HID * 4);
    float*    gsum   = (float*)   take((size_t)NG * HID * 4);
    float*    cnt    = (float*)   take((size_t)NG * 4);

    const int T = 256;
    const unsigned NEG_INF_ENC = 0x007FFFFFu;  // fenc(-inf)

    // re-init accumulators every call (deterministic work)
    hipMemsetAsync(den1, 0, (size_t)NN * HEADS * 4, stream);
    hipMemsetAsync(out1, 0, (size_t)NN * D1 * 4,    stream);
    hipMemsetAsync(den2, 0, (size_t)NN * 4,         stream);
    hipMemsetAsync(out2, 0, (size_t)NN * HID * 4,   stream);
    hipMemsetAsync(gsum, 0, (size_t)NG * HID * 4,   stream);
    hipMemsetAsync(cnt,  0, (size_t)NG * 4,         stream);
    init_u32<<<cdiv(NN*HEADS,T), T, 0, stream>>>(emax1, NEG_INF_ENC, NN*HEADS);
    init_u32<<<cdiv(NN,T),       T, 0, stream>>>(emax2, NEG_INF_ENC, NN);

    // f16 operand prep
    cvt_f32_f16<<<cdiv((size_t)NN*FIN,T), T, 0, stream>>>(x, x16, NN*FIN);
    cvt_transpose_f16<<<cdiv(D1*FIN,T),  T, 0, stream>>>(W1, W1t, D1,  FIN);
    cvt_transpose_f16<<<cdiv(HID*D1,T),  T, 0, stream>>>(W2, W2t, HID, D1);

    const int rowTiles = NN / 16;              // 1250
    const int rbBlocks = cdiv(rowTiles, 8);    // 157 (8 waves = 8 row tiles per block)

    // ---- conv1 ----
    wmma_gemm_f16<<<(D1/16) * rbBlocks, 256, 0, stream>>>(x16, W1t, hfeat1,
                                                          FIN, D1, rowTiles, rbBlocks);
    attn_dot <<<cdiv(NN*HEADS,T), T, 0, stream>>>(hfeat1, asr1, adr1, as1, ad1, HEADS);
    edge_max <<<cdiv(ETOT,T),     T, 0, stream>>>(ei, as1, ad1, emax1, HEADS);
    edge_exp <<<cdiv(ETOT,T),     T, 0, stream>>>(ei, as1, ad1, emax1, ex1, den1, HEADS);
    edge_aggr<<<cdiv((size_t)ETOT*HEADS,T), T, 0, stream>>>(ei, ex1, den1, hfeat1, out1, HEADS);
    epi1     <<<cdiv((size_t)NN*D1,T), T, 0, stream>>>(out1, b1, h1p16);

    // ---- conv2 ----
    wmma_gemm_f16<<<(HID/16) * rbBlocks, 256, 0, stream>>>(h1p16, W2t, hfeat2,
                                                           D1, HID, rowTiles, rbBlocks);
    attn_dot <<<cdiv(NN,T),   T, 0, stream>>>(hfeat2, asr2, adr2, as2, ad2, 1);
    edge_max <<<cdiv(ETOT,T), T, 0, stream>>>(ei, as2, ad2, emax2, 1);
    edge_exp <<<cdiv(ETOT,T), T, 0, stream>>>(ei, as2, ad2, emax2, ex2, den2, 1);
    edge_aggr<<<cdiv(ETOT,T), T, 0, stream>>>(ei, ex2, den2, hfeat2, out2, 1);
    epi2     <<<cdiv(NN*HID,T), T, 0, stream>>>(out2, b2, h2);

    // ---- pool + fc ----
    pool_sum<<<cdiv(NN*HID,T), T, 0, stream>>>(h2, batch, gsum, cnt);
    fc_out  <<<cdiv(NG*NCLS,T), T, 0, stream>>>(gsum, cnt, fcw, fcb, out);
}
